// RelativePositionMultiHeadAttention_11587821765320
// MI455X (gfx1250) — matrix operand order repaired and checked by reading, not compiled, not run on hardware
//
#include <hip/hip_runtime.h>
#include <hip/hip_bf16.h>

#define B_ 2
#define L_ 2048
#define E_ 512
#define H_ 8
#define D_ 64
#define RCLIP 127
#define RROWS 255
#define RPAD 256
#define INV_SCALE 0.125f
#define QT 32              // q rows per fused workgroup
#define SSTRIDE 2052       // padded LDS row stride (floats) to break bank alignment

typedef __attribute__((ext_vector_type(16))) _Float16     v16h;
typedef __attribute__((ext_vector_type(8)))  float        v8f;
typedef __attribute__((ext_vector_type(4)))  unsigned int u32x4;
typedef __attribute__((ext_vector_type(8)))  int          i32x8;
typedef __attribute__((ext_vector_type(4)))  int          i32x4;

#if __has_builtin(__builtin_amdgcn_tensor_load_to_lds)
#define HAVE_TDM 1
#else
#define HAVE_TDM 0
#endif

// ---- WMMA tile loaders (wave32, 16x16x32 f16 -> f32) -------------------
// A: 16(M) x 32(K): lane l -> m=l%16, kb=(l<16?0:8); halves [kb..kb+7],[kb+16..kb+23]
__device__ inline v16h load_a_f16(const _Float16* A, int lda, int m0, int k0, int lane) {
    const int m  = m0 + (lane & 15);
    const int kb = k0 + ((lane >> 4) << 3);
    const _Float16* p = A + (size_t)m * lda + kb;
    v16h a;
#pragma unroll
    for (int i = 0; i < 8; ++i) { a[i] = p[i]; a[8 + i] = p[16 + i]; }
    return a;
}

__device__ inline v16h load_a_f32(const float* A, int lda, int m0, int k0, int lane) {
    const int m  = m0 + (lane & 15);
    const int kb = k0 + ((lane >> 4) << 3);
    const float* p = A + (size_t)m * lda + kb;
    v16h a;
#pragma unroll
    for (int i = 0; i < 8; ++i) { a[i] = (_Float16)p[i]; a[8 + i] = (_Float16)p[16 + i]; }
    return a;
}

// A operand sourced from an LDS f32 slab (phase-3 of fused kernel).
__device__ inline v16h load_a_lds(const float* S, int lds_, int m0, int k0, int lane) {
    const int m  = m0 + (lane & 15);
    const int kb = k0 + ((lane >> 4) << 3);
    const float* p = S + (size_t)m * lds_ + kb;
    v16h a;
#pragma unroll
    for (int i = 0; i < 8; ++i) { a[i] = (_Float16)p[i]; a[8 + i] = (_Float16)p[16 + i]; }
    return a;
}

// B: 32(K) x 16(N). Source S row-major [N x K] (B = S^T): lane l -> n=l%16,
// kb=(l<16?0:8); halves [kb..kb+7],[kb+16..kb+23].
__device__ inline v16h load_b_f16(const _Float16* S, int lds_, int n0, int k0, int lane) {
    const int n  = n0 + (lane & 15);
    const int kb = k0 + ((lane >> 4) << 3);
    const _Float16* p = S + (size_t)n * lds_ + kb;
    v16h b;
#pragma unroll
    for (int i = 0; i < 8; ++i) { b[i] = p[i]; b[8 + i] = p[16 + i]; }
    return b;
}

__device__ inline v8f wmma16(v16h a, v16h b, v8f c) {
    return __builtin_amdgcn_wmma_f32_16x16x32_f16(false, a, false, b, (short)0, c, false, false);
}

// ---- K0: convert weights + rel emb to f16 ------------------------------
__global__ void convert_params(const float* Wq, const float* Wk, const float* Wv,
                               const float* Wo, const float* rel,
                               _Float16* Wqh, _Float16* Wkh, _Float16* Wvh,
                               _Float16* Woh, _Float16* relh) {
    const int EE = E_ * E_;
    const int total = 4 * EE + RPAD * D_;
    for (int i = blockIdx.x * blockDim.x + threadIdx.x; i < total; i += gridDim.x * blockDim.x) {
        if (i < EE)            Wqh[i]          = (_Float16)Wq[i];
        else if (i < 2 * EE)   Wkh[i - EE]     = (_Float16)Wk[i - EE];
        else if (i < 3 * EE)   Wvh[i - 2 * EE] = (_Float16)Wv[i - 2 * EE];
        else if (i < 4 * EE)   Woh[i - 3 * EE] = (_Float16)Wo[i - 3 * EE];
        else {
            int j = i - 4 * EE;
            relh[j] = (j < RROWS * D_) ? (_Float16)rel[j] : (_Float16)0.0f;
        }
    }
}

// ---- K1: QKV projections (x @ W^T + b), split heads --------------------
// Each wave computes a 16(m) x 64(e) strip: A loaded once per k-step, 4 WMMAs.
// Q,K stored [b,h,l,d]; V stored transposed [b,h,d,l] for the PV GEMM.
__global__ void qkv_proj(const float* qx, const float* kx, const float* vx,
                         const _Float16* Wqh, const _Float16* Wkh, const _Float16* Wvh,
                         const float* bq, const float* bk, const float* bv,
                         _Float16* Qh, _Float16* Kh, _Float16* VT) {
    const int lane = threadIdx.x;
    const int m0 = blockIdx.x * 16;
    const int e0 = blockIdx.y * 64;
    const int mode = blockIdx.z;      // 0=Q 1=K 2=V
    const float*    X = (mode == 0) ? qx  : (mode == 1) ? kx  : vx;
    const _Float16* W = (mode == 0) ? Wqh : (mode == 1) ? Wkh : Wvh;
    const float* bias = (mode == 0) ? bq  : (mode == 1) ? bk  : bv;
    _Float16*     QK  = (mode == 0) ? Qh  : Kh;

    v8f c[4] = {{}, {}, {}, {}};
    for (int kk = 0; kk < E_; kk += 32) {
        v16h a = load_a_f32(X, E_, m0, kk, lane);
#pragma unroll
        for (int j = 0; j < 4; ++j) {
            v16h b = load_b_f16(W, E_, e0 + j * 16, kk, lane);
            c[j] = wmma16(a, b, c[j]);
        }
    }
    const int roff = (lane >> 4) << 3;
#pragma unroll
    for (int j = 0; j < 4; ++j) {
        const int e = e0 + j * 16 + (lane & 15);
        const int h = e >> 6, d = e & 63;
        const float bb = bias[e];
#pragma unroll
        for (int i = 0; i < 8; ++i) {
            const int n = m0 + roff + i;
            const int bidx = n >> 11, l = n & (L_ - 1);
            const _Float16 val = (_Float16)(c[j][i] + bb);
            if (mode == 2)
                VT[((size_t)(bidx * H_ + h) * D_ + d) * L_ + l] = val;
            else
                QK[((size_t)(bidx * H_ + h) * L_ + l) * D_ + d] = val;
        }
    }
}

// ---- K2: Qrel = Q @ rel_emb^T -> [B*H, L, RPAD] f32 (16x64 per wave) ---
__global__ void qrel_gemm(const _Float16* Qh, const _Float16* relh, float* Qrel) {
    const int lane = threadIdx.x;
    const int q0 = blockIdx.x * 16;
    const int r0 = blockIdx.y * 64;
    const int bh = blockIdx.z;
    const _Float16* Qb = Qh + (size_t)bh * L_ * D_;
    v8f c[4] = {{}, {}, {}, {}};
#pragma unroll
    for (int kk = 0; kk < D_; kk += 32) {
        v16h a = load_a_f16(Qb, D_, q0, kk, lane);
#pragma unroll
        for (int j = 0; j < 4; ++j) {
            v16h b = load_b_f16(relh, D_, r0 + j * 16, kk, lane);
            c[j] = wmma16(a, b, c[j]);
        }
    }
    const int roff = (lane >> 4) << 3;
    float* O = Qrel + (size_t)bh * L_ * RPAD;
#pragma unroll
    for (int j = 0; j < 4; ++j) {
        const int r = r0 + j * 16 + (lane & 15);
#pragma unroll
        for (int i = 0; i < 8; ++i)
            O[(size_t)(q0 + roff + i) * RPAD + r] = c[j][i];
    }
}

// ---- K3: fused scores -> softmax -> attn write -> PV -------------------
// One workgroup = (bh, 32 q-rows). 8 waves. LDS: score slab [32][2052] f32,
// reductions, and a TDM-staged 32x64 f16 Q tile.
__global__ void fused_attn(const _Float16* Qh, const _Float16* Kh,
                           const float* Qrel, const _Float16* VT,
                           float* attn, _Float16* ctx) {
    extern __shared__ float smem[];
    float*     sc     = smem;                       // [QT][SSTRIDE]
    float*     redmax = sc + QT * SSTRIDE;          // [QT][8]
    float*     redsum = redmax + QT * 8;            // [QT][8]
    float*     invrow = redsum + QT * 8;            // [QT]
    _Float16*  sQ     = (_Float16*)(invrow + QT);   // [QT][D_]

    const int t    = threadIdx.x;
    const int lane = t & 31;
    const int wv   = t >> 5;                        // wave 0..7
    const int q0   = blockIdx.x * QT;
    const int bh   = blockIdx.y;
    const int bidx = bh >> 3, h = bh & (H_ - 1);

    const _Float16* Qb = Qh   + (size_t)bh * L_ * D_;
    const _Float16* Kb = Kh   + (size_t)bh * L_ * D_;
    const float*    Rb = Qrel + (size_t)bh * L_ * RPAD;
    const _Float16* Vb = VT   + (size_t)bh * D_ * L_;
    float*          Ab = attn + (size_t)bh * L_ * L_;

    // ---- Stage the 32x64 f16 Q tile into LDS via the Tensor Data Mover.
#if HAVE_TDM
    if (wv == 0) {
        const unsigned long long ga = (unsigned long long)(size_t)(Qb + (size_t)q0 * D_);
        const unsigned lofs = (unsigned)(size_t)sQ;   // LDS aperture: low 32 bits = LDS offset
        u32x4 g0;
        g0[0] = 1u;                                   // count=1, user mode
        g0[1] = lofs;                                 // lds_addr
        g0[2] = (unsigned)(ga & 0xffffffffu);         // global_addr[31:0]
        g0[3] = (unsigned)((ga >> 32) & 0x01ffffffu)  // global_addr[56:32]
              | (2u << 30);                           // type=2 ("image")
        i32x8 g1;
        g1[0] = 0x00010000;          // wg_mask=0, data_size=1 (2 bytes)
        g1[1] = (int)(64u << 16);    // atomic_barrier=0, tensor_dim0[15:0]=64
        g1[2] = (int)(2048u << 16);  // tensor_dim0[31:16]=0, tensor_dim1[15:0]=2048
        g1[3] = (int)(64u << 16);    // tensor_dim1[31:16]=0, tile_dim0=64
        g1[4] = 32;                  // tile_dim1=32, tile_dim2=0
        g1[5] = 64;                  // tensor_dim0_stride[31:0]=64
        g1[6] = 0;                   // stride hi / dim1_stride lo
        g1[7] = 0;
        i32x4 gz = {0, 0, 0, 0};
#if defined(__clang_major__) && __clang_major__ >= 23
        i32x8 gz8 = {0, 0, 0, 0, 0, 0, 0, 0};
        __builtin_amdgcn_tensor_load_to_lds(g0, g1, gz, gz, gz8, 0);
#else
        __builtin_amdgcn_tensor_load_to_lds(g0, g1, gz, gz, 0);
#endif
        __builtin_amdgcn_s_wait_tensorcnt(0);
    }
#else
    for (int idx = t; idx < QT * D_; idx += 256)
        sQ[idx] = Qb[(size_t)(q0 + (idx >> 6)) * D_ + (idx & 63)];
#endif
    __syncthreads();

    // ---- Phase 1: logits for 32 q-rows x 2048 k-cols into LDS ----------
    const int kc0  = wv * 256;                      // this wave's k chunk
    const int roff = (lane >> 4) << 3;
    {
        // Q A-operands for both 16-row subtiles, cached in VGPRs (from LDS).
        const v16h a00 = load_a_f16(sQ, D_, 0,  0,  lane);
        const v16h a01 = load_a_f16(sQ, D_, 0,  32, lane);
        const v16h a10 = load_a_f16(sQ, D_, 16, 0,  lane);
        const v16h a11 = load_a_f16(sQ, D_, 16, 32, lane);
        for (int kt = 0; kt < 16; ++kt) {
            const int kcol = kc0 + kt * 16;
            const v16h b0 = load_b_f16(Kb, D_, kcol, 0,  lane);
            const v16h b1 = load_b_f16(Kb, D_, kcol, 32, lane);
            const int k = kcol + (lane & 15);
#pragma unroll
            for (int qsub = 0; qsub < 2; ++qsub) {
                v8f c = {};
                c = wmma16(qsub ? a10 : a00, b0, c);
                c = wmma16(qsub ? a11 : a01, b1, c);
#pragma unroll
                for (int i = 0; i < 8; ++i) {
                    const int row = qsub * 16 + roff + i;
                    const int q = q0 + row;
                    int rel = k - q;
                    rel = rel < -RCLIP ? -RCLIP : (rel > RCLIP ? RCLIP : rel);
                    sc[row * SSTRIDE + k] =
                        (c[i] + Rb[(size_t)q * RPAD + (rel + RCLIP)]) * INV_SCALE;
                }
            }
        }
    }
    __syncthreads();

    // ---- Phase 2: row softmax (8 threads per row) ----------------------
    {
        const int r = t >> 3, s = t & 7;
        const float* p = sc + r * SSTRIDE + s * 256;
        float m = -1e30f;
        for (int j = 0; j < 256; ++j) m = fmaxf(m, p[j]);
        redmax[r * 8 + s] = m;
    }
    __syncthreads();
    {
        const int r = t >> 3, s = t & 7;
        float m = redmax[r * 8];
#pragma unroll
        for (int j = 1; j < 8; ++j) m = fmaxf(m, redmax[r * 8 + j]);
        float* p = sc + r * SSTRIDE + s * 256;
        float acc = 0.f;
        for (int j = 0; j < 256; ++j) {
            float e = __expf(p[j] - m);
            p[j] = e;
            acc += e;
        }
        redsum[r * 8 + s] = acc;
    }
    __syncthreads();
    if ((t & 7) == 0) {
        const int r = t >> 3;
        float sum = redsum[r * 8];
#pragma unroll
        for (int j = 1; j < 8; ++j) sum += redsum[r * 8 + j];
        invrow[r] = 1.0f / sum;
    }
    __syncthreads();
    // Normalize in LDS and write attn_weights (coalesced: lanes sweep k).
    for (int idx = t; idx < QT * L_; idx += 256) {
        const int row = idx >> 11, k = idx & (L_ - 1);
        const float pv = sc[row * SSTRIDE + k] * invrow[row];
        sc[row * SSTRIDE + k] = pv;
        Ab[(size_t)(q0 + row) * L_ + k] = pv;
    }
    __syncthreads();

    // ---- Phase 3: ctx = P @ V from LDS probs ---------------------------
    {
        const int qsub = wv >> 2;               // 0..1
        const int d0   = (wv & 3) * 16;         // 0..48
        v8f c = {};
        for (int kk = 0; kk < L_; kk += 32) {
            __builtin_prefetch(Vb + (size_t)(d0 + (lane & 15)) * L_ + kk + 64, 0, 1);
            v16h a = load_a_lds(sc, SSTRIDE, qsub * 16, kk, lane);
            v16h b = load_b_f16(Vb, L_, d0, kk, lane);
            c = wmma16(a, b, c);
        }
        const int e = h * D_ + d0 + (lane & 15);
#pragma unroll
        for (int i = 0; i < 8; ++i) {
            const int q = q0 + qsub * 16 + roff + i;
            ctx[(size_t)(bidx * L_ + q) * E_ + e] = (_Float16)c[i];
        }
    }
}

// ---- K4: out = ctx @ Wo^T + bo (16x64 strip per wave) ------------------
__global__ void out_proj(const _Float16* ctx, const _Float16* Woh, const float* bo, float* out) {
    const int lane = threadIdx.x;
    const int m0 = blockIdx.x * 16;
    const int e0 = blockIdx.y * 64;
    v8f c[4] = {{}, {}, {}, {}};
    for (int kk = 0; kk < E_; kk += 32) {
        v16h a = load_a_f16(ctx, E_, m0, kk, lane);
#pragma unroll
        for (int j = 0; j < 4; ++j) {
            v16h b = load_b_f16(Woh, E_, e0 + j * 16, kk, lane);
            c[j] = wmma16(a, b, c[j]);
        }
    }
    const int roff = (lane >> 4) << 3;
#pragma unroll
    for (int j = 0; j < 4; ++j) {
        const int e = e0 + j * 16 + (lane & 15);
        const float bb = bo[e];
#pragma unroll
        for (int i = 0; i < 8; ++i)
            out[(size_t)(m0 + roff + i) * E_ + e] = c[j][i] + bb;
    }
}

extern "C" void kernel_launch(void* const* d_in, const int* in_sizes, int n_in,
                              void* d_out, int out_size, void* d_ws, size_t ws_size,
                              hipStream_t stream) {
    const float* qx  = (const float*)d_in[0];
    const float* kx  = (const float*)d_in[1];
    const float* vx  = (const float*)d_in[2];
    const float* Wq  = (const float*)d_in[3];
    const float* bq  = (const float*)d_in[4];
    const float* Wk  = (const float*)d_in[5];
    const float* bk  = (const float*)d_in[6];
    const float* Wv  = (const float*)d_in[7];
    const float* bv  = (const float*)d_in[8];
    const float* Wo  = (const float*)d_in[9];
    const float* bo  = (const float*)d_in[10];
    const float* rel = (const float*)d_in[11];

    float* out  = (float*)d_out;
    float* attn = out + (size_t)B_ * L_ * E_;

    char* ws = (char*)d_ws;
    size_t off = 0;
    auto carve = [&](size_t bytes) -> void* {
        void* p = ws + off;
        off += (bytes + 255) & ~(size_t)255;
        return p;
    };
    _Float16* Qh   = (_Float16*)carve((size_t)B_ * H_ * L_ * D_ * sizeof(_Float16));
    _Float16* Kh   = (_Float16*)carve((size_t)B_ * H_ * L_ * D_ * sizeof(_Float16));
    _Float16* VT   = (_Float16*)carve((size_t)B_ * H_ * L_ * D_ * sizeof(_Float16));
    _Float16* Wqh  = (_Float16*)carve((size_t)E_ * E_ * sizeof(_Float16));
    _Float16* Wkh  = (_Float16*)carve((size_t)E_ * E_ * sizeof(_Float16));
    _Float16* Wvh  = (_Float16*)carve((size_t)E_ * E_ * sizeof(_Float16));
    _Float16* Woh  = (_Float16*)carve((size_t)E_ * E_ * sizeof(_Float16));
    _Float16* relh = (_Float16*)carve((size_t)RPAD * D_ * sizeof(_Float16));
    _Float16* ctx  = (_Float16*)carve((size_t)B_ * L_ * E_ * sizeof(_Float16));
    float*    Qrel = (float*)   carve((size_t)B_ * H_ * L_ * RPAD * sizeof(float));

    convert_params<<<512, 256, 0, stream>>>(Wq, Wk, Wv, Wo, rel, Wqh, Wkh, Wvh, Woh, relh);

    qkv_proj<<<dim3((B_ * L_) / 16, E_ / 64, 3), 32, 0, stream>>>(
        qx, kx, vx, Wqh, Wkh, Wvh, bq, bk, bv, Qh, Kh, VT);

    qrel_gemm<<<dim3(L_ / 16, RPAD / 64, B_ * H_), 32, 0, stream>>>(Qh, relh, Qrel);

    const size_t lds_bytes = (size_t)(QT * SSTRIDE + QT * 8 * 2 + QT) * sizeof(float)
                           + (size_t)QT * D_ * sizeof(_Float16);
    fused_attn<<<dim3(L_ / QT, B_ * H_), 256, lds_bytes, stream>>>(
        Qh, Kh, Qrel, VT, attn, ctx);

    out_proj<<<dim3((B_ * L_) / 16, E_ / 64), 32, 0, stream>>>(ctx, Woh, bo, out);
}
